// selfattentionlayer_nat_54735063220680
// MI455X (gfx1250) — matrix operand order repaired and checked by reading, not compiled, not run on hardware
//
#include <hip/hip_runtime.h>
#include <hip/hip_bf16.h>

typedef __bf16 bf16_t;
typedef __attribute__((ext_vector_type(16))) __bf16 v16bf;
typedef __attribute__((ext_vector_type(8)))  __bf16 v8bf;
typedef __attribute__((ext_vector_type(8)))  float  v8f;

#define NB      16        // batch
#define HH      64
#define WW      64
#define HW      4096
#define CC      128
#define VC      64
#define EC      256
#define AA      128
#define NHEAD   8
#define HD      16
#define FF      512
#define KWIN    11
#define NTOK    (NB*HW)   // 65536
#define NTILE   (NTOK/16) // 4096
#define SCALE_Q 0.25f

// ---------------- workspace layout (bytes) ----------------
#define OFF_WCTXP 0u                      // 256x64  bf16 (transposed NxK)
#define OFF_WQ    32768u                  // 192x128 bf16
#define OFF_WK    81920u
#define OFF_WV    131072u                 // 128x128
#define OFF_WP    163840u                 // 256x128
#define OFF_W1    229376u                 // 128x512
#define OFF_W2    360448u                 // 512x128
#define OFF_CP    491520u                 // ctx proj f32  NTOK*64
#define OFF_Q     17268736u               // f32 NTOK*128
#define OFF_K     50823168u
#define OFF_V     84377600u
#define OFF_O     117932032u
#define OFF_Y     151486464u

// ---------------- WMMA fragment helpers (A/B use contiguous k order; D per ISA layout) -------------
// A (16x32 bf16): lane row = lane&15 ; lane holds K = 16*hi .. 16*hi+15
// contiguous  (hi = lane>>4)
__device__ __forceinline__ v16bf lda(const bf16_t* Abuf, int ldk, int kbase, int lane) {
    int row = lane & 15, hi = lane >> 4;
    return *(const v16bf*)(Abuf + row * ldk + kbase + hi * 16);
}
// B (32x16 bf16): lane holds column N = lane&15, K = 16*hi .. 16*hi+15
// contiguous.  Weights stored transposed: Wt[n*Kd + k].
__device__ __forceinline__ v16bf ldb(const bf16_t* Wt, int Kd, int kbase, int n0, int lane) {
    return *(const v16bf*)(Wt + (size_t)(n0 + (lane & 15)) * Kd + kbase + ((lane >> 4) << 4));
}
// D/C (16x16 f32): c[r] = (M = r + 8*(lane>>4), N = lane&15)
__device__ __forceinline__ void std_store(float* out, int ldn, const v8f& c, int lane) {
    int n = lane & 15, mb = (lane >> 4) * 8;
#pragma unroll
    for (int r = 0; r < 8; ++r) out[(size_t)(mb + r) * ldn + n] = c[r];
}

#define WMMA_BF16(acc, a, b) \
    acc = __builtin_amdgcn_wmma_f32_16x16x32_bf16(false, (a), false, (b), (short)0, (acc), false, false)

__device__ __forceinline__ void wred2(float& s, float& ss) {
#pragma unroll
    for (int o = 16; o >= 1; o >>= 1) {
        s  += __shfl_xor(s, o, 32);
        ss += __shfl_xor(ss, o, 32);
    }
}

// ---------------- kernel 0: f32 -> bf16 transposed weight convert -----------
__global__ void conv_t_kernel(const float* __restrict__ w, bf16_t* __restrict__ wt,
                              int Kd, int Nd) {
    int i = blockIdx.x * 256 + threadIdx.x;
    if (i < Kd * Nd) {
        int k = i / Nd, n = i - k * Nd;
        wt[(size_t)n * Kd + k] = (bf16_t)w[i];
    }
}

// ---------------- kernel 1: ctx LN + 256->64 projection ---------------------
__global__ void ctx_proj_kernel(const float* __restrict__ ctx,
                                const float* __restrict__ g, const float* __restrict__ be,
                                const bf16_t* __restrict__ wt, const float* __restrict__ pb,
                                float* __restrict__ out) {
    __shared__ __align__(32) bf16_t sA[4][16 * 256];
    int wave = threadIdx.x >> 5, lane = threadIdx.x & 31;
    int tile = blockIdx.x * 4 + wave;
    int tok0 = tile * 16;
    int b = tok0 >> 12, pix0 = tok0 & 4095;
    bf16_t* A = sA[wave];
    for (int i = 0; i < 16; ++i) {
        const float* base = ctx + ((size_t)b * EC) * HW + (pix0 + i);
        float v[8], s = 0.f, ss = 0.f;
#pragma unroll
        for (int j = 0; j < 8; ++j) {
            int c = lane + j * 32;
            v[j] = base[(size_t)c * HW];
            s += v[j]; ss += v[j] * v[j];
        }
        wred2(s, ss);
        float mu = s * (1.0f / 256.0f);
        float rs = rsqrtf(ss * (1.0f / 256.0f) - mu * mu + 1e-5f);
#pragma unroll
        for (int j = 0; j < 8; ++j) {
            int c = lane + j * 32;
            A[i * 256 + c] = (bf16_t)((v[j] - mu) * rs * g[c] + be[c]);
        }
    }
    __syncthreads();
    for (int n0 = 0; n0 < 64; n0 += 16) {
        v8f acc; float bias = pb[n0 + (lane & 15)];
#pragma unroll
        for (int r = 0; r < 8; ++r) acc[r] = bias;
#pragma unroll
        for (int kb = 0; kb < 8; ++kb) {
            v16bf a = lda(A, 256, kb * 32, lane);
            v16bf bm = ldb(wt, 256, kb * 32, n0, lane);
            WMMA_BF16(acc, a, bm);
        }
        std_store(out + (size_t)tok0 * 64 + n0, 64, acc, lane);
    }
}

// ---------------- kernel 2: LNs + Q/K/V GEMMs -------------------------------
__global__ void qkv_kernel(const float* __restrict__ x, const float* __restrict__ cp,
                           const float* __restrict__ n1g, const float* __restrict__ n1b,
                           const float* __restrict__ nvg, const float* __restrict__ nvb,
                           const bf16_t* __restrict__ wq, const float* __restrict__ qb,
                           const bf16_t* __restrict__ wk, const float* __restrict__ kb2,
                           const bf16_t* __restrict__ wv, const float* __restrict__ vb,
                           float* __restrict__ q, float* __restrict__ k, float* __restrict__ v) {
    __shared__ __align__(32) bf16_t sQK[4][16 * 192];
    __shared__ __align__(32) bf16_t sVV[4][16 * 128];
    int wave = threadIdx.x >> 5, lane = threadIdx.x & 31;
    int tile = blockIdx.x * 4 + wave;
    int tok0 = tile * 16;
    bf16_t* AQK = sQK[wave];
    bf16_t* AVV = sVV[wave];
    for (int i = 0; i < 16; ++i) {
        int tok = tok0 + i;
        const float* xr = x + (size_t)tok * CC;
        const float* cr = cp + (size_t)tok * VC;
        // LN over 192 = [x | ctx]
        float v6[6], s = 0.f, ss = 0.f;
#pragma unroll
        for (int j = 0; j < 6; ++j) {
            int c = lane + j * 32;
            v6[j] = (c < CC) ? xr[c] : cr[c - CC];
            s += v6[j]; ss += v6[j] * v6[j];
        }
        wred2(s, ss);
        float mu = s * (1.0f / 192.0f);
        float rs = rsqrtf(ss * (1.0f / 192.0f) - mu * mu + 1e-5f);
#pragma unroll
        for (int j = 0; j < 6; ++j) {
            int c = lane + j * 32;
            AQK[i * 192 + c] = (bf16_t)((v6[j] - mu) * rs * n1g[c] + n1b[c]);
        }
        // LN over 128 = x
        float v4[4]; s = 0.f; ss = 0.f;
#pragma unroll
        for (int j = 0; j < 4; ++j) {
            int c = lane + j * 32;
            v4[j] = xr[c];
            s += v4[j]; ss += v4[j] * v4[j];
        }
        wred2(s, ss);
        mu = s * (1.0f / 128.0f);
        rs = rsqrtf(ss * (1.0f / 128.0f) - mu * mu + 1e-5f);
#pragma unroll
        for (int j = 0; j < 4; ++j) {
            int c = lane + j * 32;
            AVV[i * 128 + c] = (bf16_t)((v4[j] - mu) * rs * nvg[c] + nvb[c]);
        }
    }
    __syncthreads();
    // Q (K=192) with scale
    for (int n0 = 0; n0 < 128; n0 += 16) {
        v8f acc; float bias = qb[n0 + (lane & 15)];
#pragma unroll
        for (int r = 0; r < 8; ++r) acc[r] = bias;
#pragma unroll
        for (int kb = 0; kb < 6; ++kb) {
            v16bf a = lda(AQK, 192, kb * 32, lane);
            v16bf bm = ldb(wq, 192, kb * 32, n0, lane);
            WMMA_BF16(acc, a, bm);
        }
#pragma unroll
        for (int r = 0; r < 8; ++r) acc[r] *= SCALE_Q;
        std_store(q + (size_t)tok0 * AA + n0, AA, acc, lane);
    }
    // K (K=192)
    for (int n0 = 0; n0 < 128; n0 += 16) {
        v8f acc; float bias = kb2[n0 + (lane & 15)];
#pragma unroll
        for (int r = 0; r < 8; ++r) acc[r] = bias;
#pragma unroll
        for (int kb = 0; kb < 6; ++kb) {
            v16bf a = lda(AQK, 192, kb * 32, lane);
            v16bf bm = ldb(wk, 192, kb * 32, n0, lane);
            WMMA_BF16(acc, a, bm);
        }
        std_store(k + (size_t)tok0 * AA + n0, AA, acc, lane);
    }
    // V (K=128)
    for (int n0 = 0; n0 < 128; n0 += 16) {
        v8f acc; float bias = vb[n0 + (lane & 15)];
#pragma unroll
        for (int r = 0; r < 8; ++r) acc[r] = bias;
#pragma unroll
        for (int kb = 0; kb < 4; ++kb) {
            v16bf a = lda(AVV, 128, kb * 32, lane);
            v16bf bm = ldb(wv, 128, kb * 32, n0, lane);
            WMMA_BF16(acc, a, bm);
        }
        std_store(v + (size_t)tok0 * AA + n0, AA, acc, lane);
    }
}

// ---------------- kernel 3: neighborhood attention (online softmax) ---------
__global__ void nat_kernel(const float* __restrict__ qg, const float* __restrict__ kg,
                           const float* __restrict__ vg, const float* __restrict__ rpb,
                           float* __restrict__ og) {
    int idx = blockIdx.x * 256 + threadIdx.x;   // < NB*NHEAD*HW
    int xx = idx & 63;
    int yy = (idx >> 6) & 63;
    int h  = (idx >> 12) & 7;
    int b  = idx >> 15;
    size_t base = ((size_t)b * HW + yy * WW + xx) * AA + h * HD;
    float qv[16];
#pragma unroll
    for (int d = 0; d < 16; ++d) qv[d] = qg[base + d];
    int iy0 = min(max(yy - 5, 0), HH - KWIN);
    int ix0 = min(max(xx - 5, 0), WW - KWIN);
    const float* bias = rpb + h * (2 * KWIN - 1) * (2 * KWIN - 1);
    float m = -3.0e38f, l = 0.f;
    float acc[16];
#pragma unroll
    for (int d = 0; d < 16; ++d) acc[d] = 0.f;
    for (int a = 0; a < KWIN; ++a) {
        int ry = iy0 + a;
        const float* krow = kg + ((size_t)b * HW + ry * WW) * AA + h * HD;
        const float* vrow = vg + ((size_t)b * HW + ry * WW) * AA + h * HD;
        int bb = (ry - yy + (KWIN - 1)) * (2 * KWIN - 1) + (ix0 - xx + (KWIN - 1));
        for (int c = 0; c < KWIN; ++c) {
            const float* kp = krow + (size_t)(ix0 + c) * AA;
            float s = 0.f;
#pragma unroll
            for (int d = 0; d < 16; ++d) s = fmaf(qv[d], kp[d], s);
            s += bias[bb + c];
            float mn   = fmaxf(m, s);
            float corr = __expf(m - mn);
            float p    = __expf(s - mn);
            l = l * corr + p;
            const float* vp = vrow + (size_t)(ix0 + c) * AA;
#pragma unroll
            for (int d = 0; d < 16; ++d) acc[d] = fmaf(acc[d], corr, p * vp[d]);
            m = mn;
        }
    }
    float inv = 1.0f / l;
#pragma unroll
    for (int d = 0; d < 16; ++d) og[base + d] = acc[d] * inv;
}

// ---------------- kernel 4: output projection + residual --------------------
__global__ void proj_kernel(const float* __restrict__ o, const float* __restrict__ x,
                            const bf16_t* __restrict__ wt, const float* __restrict__ pb,
                            float* __restrict__ y) {
    __shared__ __align__(32) bf16_t sA[4][16 * 256];
    int wave = threadIdx.x >> 5, lane = threadIdx.x & 31;
    int tile = blockIdx.x * 4 + wave;
    int tok0 = tile * 16;
    bf16_t* A = sA[wave];
    for (int i = 0; i < 16; ++i) {
        int tok = tok0 + i;
#pragma unroll
        for (int j = 0; j < 8; ++j) {
            int c = lane + j * 32;
            float val = (c < CC) ? o[(size_t)tok * AA + c] : x[(size_t)tok * CC + (c - CC)];
            A[i * 256 + c] = (bf16_t)val;
        }
    }
    __syncthreads();
    for (int n0 = 0; n0 < 128; n0 += 16) {
        v8f acc; float bias = pb[n0 + (lane & 15)];
#pragma unroll
        for (int r = 0; r < 8; ++r) acc[r] = bias;
#pragma unroll
        for (int kb = 0; kb < 8; ++kb) {
            v16bf a = lda(A, 256, kb * 32, lane);
            v16bf bm = ldb(wt, 256, kb * 32, n0, lane);
            WMMA_BF16(acc, a, bm);
        }
        int n = lane & 15, mb = (lane >> 4) * 8;
#pragma unroll
        for (int r = 0; r < 8; ++r) {
            size_t ro = (size_t)(tok0 + mb + r) * CC + n0 + n;
            y[ro] = x[ro] + acc[r];
        }
    }
}

// ---------------- kernel 5: FFN (LN -> GEMM -> GELU -> GEMM -> residual) ----
__global__ void ffn_kernel(const float* __restrict__ y,
                           const float* __restrict__ g, const float* __restrict__ be,
                           const bf16_t* __restrict__ w1, const float* __restrict__ b1,
                           const bf16_t* __restrict__ w2, const float* __restrict__ b2,
                           float* __restrict__ out) {
    __shared__ __align__(32) bf16_t sA[2][16 * 128];
    __shared__ __align__(32) bf16_t sH[2][16 * 512];
    int wave = threadIdx.x >> 5, lane = threadIdx.x & 31;
    int tile = blockIdx.x * 2 + wave;
    int tok0 = tile * 16;
    bf16_t* A = sA[wave];
    bf16_t* Hd = sH[wave];
    for (int i = 0; i < 16; ++i) {
        const float* yr = y + (size_t)(tok0 + i) * CC;
        float v4[4], s = 0.f, ss = 0.f;
#pragma unroll
        for (int j = 0; j < 4; ++j) {
            int c = lane + j * 32;
            v4[j] = yr[c];
            s += v4[j]; ss += v4[j] * v4[j];
        }
        wred2(s, ss);
        float mu = s * (1.0f / 128.0f);
        float rs = rsqrtf(ss * (1.0f / 128.0f) - mu * mu + 1e-5f);
#pragma unroll
        for (int j = 0; j < 4; ++j) {
            int c = lane + j * 32;
            A[i * 128 + c] = (bf16_t)((v4[j] - mu) * rs * g[c] + be[c]);
        }
    }
    __syncthreads();
    // GEMM1 128->512 + exact GELU -> LDS (bf16)
    int n = lane & 15, mb = (lane >> 4) * 8;
    for (int n0 = 0; n0 < 512; n0 += 16) {
        v8f acc; float bias = b1[n0 + n];
#pragma unroll
        for (int r = 0; r < 8; ++r) acc[r] = bias;
#pragma unroll
        for (int kb = 0; kb < 4; ++kb) {
            v16bf a = lda(A, 128, kb * 32, lane);
            v16bf bm = ldb(w1, 128, kb * 32, n0, lane);
            WMMA_BF16(acc, a, bm);
        }
#pragma unroll
        for (int r = 0; r < 8; ++r) {
            float hv = acc[r];
            float gel = 0.5f * hv * (1.0f + erff(hv * 0.70710678118654752f));
            Hd[(mb + r) * 512 + n0 + n] = (bf16_t)gel;
        }
    }
    __syncthreads();
    // GEMM2 512->128 + residual
    for (int n0 = 0; n0 < 128; n0 += 16) {
        v8f acc; float bias = b2[n0 + n];
#pragma unroll
        for (int r = 0; r < 8; ++r) acc[r] = bias;
#pragma unroll
        for (int kb = 0; kb < 16; ++kb) {
            v16bf a = lda(Hd, 512, kb * 32, lane);
            v16bf bm = ldb(w2, 512, kb * 32, n0, lane);
            WMMA_BF16(acc, a, bm);
        }
#pragma unroll
        for (int r = 0; r < 8; ++r) {
            size_t ro = (size_t)(tok0 + mb + r) * CC + n0 + n;
            out[ro] = y[ro] + acc[r];
        }
    }
}

// ---------------- launch ----------------------------------------------------
extern "C" void kernel_launch(void* const* d_in, const int* in_sizes, int n_in,
                              void* d_out, int out_size, void* d_ws, size_t ws_size,
                              hipStream_t stream) {
    const float* x      = (const float*)d_in[0];
    const float* contex = (const float*)d_in[1];
    const float* ctxn_g = (const float*)d_in[2];
    const float* ctxn_b = (const float*)d_in[3];
    const float* ctxp_w = (const float*)d_in[4];
    const float* ctxp_b = (const float*)d_in[5];
    const float* n1_g   = (const float*)d_in[6];
    const float* n1_b   = (const float*)d_in[7];
    const float* n1v_g  = (const float*)d_in[8];
    const float* n1v_b  = (const float*)d_in[9];
    const float* q_w    = (const float*)d_in[10];
    const float* q_b    = (const float*)d_in[11];
    const float* k_w    = (const float*)d_in[12];
    const float* k_b    = (const float*)d_in[13];
    const float* v_w    = (const float*)d_in[14];
    const float* v_b    = (const float*)d_in[15];
    const float* rpb    = (const float*)d_in[16];
    const float* proj_w = (const float*)d_in[17];
    const float* proj_b = (const float*)d_in[18];
    const float* n2_g   = (const float*)d_in[19];
    const float* n2_b   = (const float*)d_in[20];
    const float* f1_w   = (const float*)d_in[21];
    const float* f1_b   = (const float*)d_in[22];
    const float* f2_w   = (const float*)d_in[23];
    const float* f2_b   = (const float*)d_in[24];
    float* out = (float*)d_out;

    char* ws = (char*)d_ws;
    bf16_t* wt_ctxp = (bf16_t*)(ws + OFF_WCTXP);
    bf16_t* wt_q    = (bf16_t*)(ws + OFF_WQ);
    bf16_t* wt_k    = (bf16_t*)(ws + OFF_WK);
    bf16_t* wt_v    = (bf16_t*)(ws + OFF_WV);
    bf16_t* wt_p    = (bf16_t*)(ws + OFF_WP);
    bf16_t* wt_1    = (bf16_t*)(ws + OFF_W1);
    bf16_t* wt_2    = (bf16_t*)(ws + OFF_W2);
    float* cp = (float*)(ws + OFF_CP);
    float* q  = (float*)(ws + OFF_Q);
    float* k  = (float*)(ws + OFF_K);
    float* v  = (float*)(ws + OFF_V);
    float* o  = (float*)(ws + OFF_O);
    float* y  = (float*)(ws + OFF_Y);

    // weight conversion (f32 -> bf16, transposed to N-major)
    conv_t_kernel<<<(EC * VC + 255) / 256, 256, 0, stream>>>(ctxp_w, wt_ctxp, EC, VC);
    conv_t_kernel<<<(192 * 128 + 255) / 256, 256, 0, stream>>>(q_w, wt_q, 192, 128);
    conv_t_kernel<<<(192 * 128 + 255) / 256, 256, 0, stream>>>(k_w, wt_k, 192, 128);
    conv_t_kernel<<<(128 * 128 + 255) / 256, 256, 0, stream>>>(v_w, wt_v, 128, 128);
    conv_t_kernel<<<(256 * 128 + 255) / 256, 256, 0, stream>>>(proj_w, wt_p, 256, 128);
    conv_t_kernel<<<(128 * 512 + 255) / 256, 256, 0, stream>>>(f1_w, wt_1, 128, 512);
    conv_t_kernel<<<(512 * 128 + 255) / 256, 256, 0, stream>>>(f2_w, wt_2, 512, 128);

    ctx_proj_kernel<<<NTILE / 4, 128, 0, stream>>>(contex, ctxn_g, ctxn_b, wt_ctxp, ctxp_b, cp);
    qkv_kernel<<<NTILE / 4, 128, 0, stream>>>(x, cp, n1_g, n1_b, n1v_g, n1v_b,
                                              wt_q, q_b, wt_k, k_b, wt_v, v_b, q, k, v);
    nat_kernel<<<(NB * NHEAD * HW) / 256, 256, 0, stream>>>(q, k, v, rpb, o);
    proj_kernel<<<NTILE / 4, 128, 0, stream>>>(o, x, wt_p, proj_b, y);
    ffn_kernel<<<NTILE / 2, 64, 0, stream>>>(y, n2_g, n2_b, wt_1, f1_b, wt_2, f2_b, out);

    (void)in_sizes; (void)n_in; (void)out_size; (void)ws_size;
}
